// AdjPolicy_82617990906012
// MI455X (gfx1250) — matrix operand order repaired and checked by reading, not compiled, not run on hardware
//
#include <hip/hip_runtime.h>
#include <hip/hip_bf16.h>

// ---------------------------------------------------------------------------
// AdjPolicy hypernet mixer for MI455X (gfx1250, wave32, WMMA).
// Heavy lifting = 7 dense GEMMs (620 GFLOP) -> bf16 WMMA 16x16x32, f32 acc,
// with GLOBAL_LOAD_ASYNC_TO_LDS_B128 double-buffered staging (ASYNCcnt).
// Tiny per-sample mixing (4 GFLOP) -> f32 VALU kernel (also async staging).
// ---------------------------------------------------------------------------

typedef __bf16  v16bf __attribute__((ext_vector_type(16)));
typedef float   v8f   __attribute__((ext_vector_type(8)));

#define B_SZ   16384
#define NV_SZ  32
#define NF_SZ  64
#define AOD_SZ 64
#define SD_SZ  512
#define H_SZ   2048   // AOD*AOD/2

__device__ __forceinline__ unsigned short f2bf(float f) {
    unsigned int u = __float_as_uint(f);
    unsigned int r = u + 0x7FFFu + ((u >> 16) & 1u);   // round to nearest even
    return (unsigned short)(r >> 16);
}

// LDS byte address of a __shared__ object: flat LDS pointers carry the LDS
// offset in addr[31:0] (ISA 10.2 aperture rule).
__device__ __forceinline__ unsigned int lds_addr(const void* p) {
    return (unsigned int)(uintptr_t)p;
}

// Async DMA: 16 bytes per lane, global -> LDS, tracked by ASYNCcnt.
__device__ __forceinline__ void async_b128(unsigned int lds, const void* gaddr) {
    asm volatile("global_load_async_to_lds_b128 %0, %1, off"
                 :: "v"(lds), "v"(gaddr) : "memory");
}
__device__ __forceinline__ void wait_async0() {
    asm volatile("s_wait_asynccnt 0x0" ::: "memory");
}

// -------------------------- conversion kernels -----------------------------

__global__ void cvt_bf16_kernel(const float* __restrict__ src,
                                unsigned short* __restrict__ dst, int n) {
    int i = blockIdx.x * 256 + threadIdx.x;
    if (i < n) dst[i] = f2bf(src[i]);
}

// Wt[n*K + k] = bf16(W[k*N + n])  (row-major [K,N] f32 -> row-major [N,K] bf16)
__global__ void cvt_transpose_kernel(const float* __restrict__ W,
                                     unsigned short* __restrict__ Wt,
                                     int K, int N) {
    int i = blockIdx.x * 256 + threadIdx.x;
    if (i < K * N) {
        int n = i / K;
        int k = i - n * K;
        Wt[(size_t)n * K + k] = f2bf(W[(size_t)k * N + n]);
    }
}

// ------------------------------ WMMA GEMM ----------------------------------
// C[M,N] = act(A[M,K] @ Bt[N,K]^T + bias[N]) ; A,Bt bf16 ; C bf16 or f32.
// Block: 256 threads (8 waves). Tile: 128(M) x 128(N), K-step 32.
// Wave (wm in 0..3, wn in 0..1) computes a 32x64 patch = 2x4 WMMA tiles.
// LDS double buffer filled by async global->LDS DMA.

#define LDS_PITCH 40   // shorts per tile row (32 data + 8 pad)

template <bool RELU, bool OUT_BF16>
__global__ __launch_bounds__(256)
void gemm_bf16_wmma(const unsigned short* __restrict__ A,
                    const unsigned short* __restrict__ Bt,
                    const float* __restrict__ bias,
                    void* __restrict__ Out,
                    int N, int K) {
    __shared__ __align__(16) unsigned short As[2][128 * LDS_PITCH];
    __shared__ __align__(16) unsigned short Bs[2][128 * LDS_PITCH];

    const int t    = threadIdx.x;
    const int lane = t & 31;
    const int wave = t >> 5;
    const int wm   = wave & 3;   // 0..3  -> M sub-block of 32
    const int wn   = wave >> 2;  // 0..1  -> N sub-block of 64
    const int m0   = blockIdx.y * 128;
    const int n0   = blockIdx.x * 128;

    // staging: 512 x 16B chunks per 128x32 tile, 2 A + 2 B chunks per thread
    const int r0  = t >> 2;          // chunk row (0..63), +64 for second chunk
    const int off = (t & 3) * 8;     // short offset within 32-short row

    auto issueTile = [&](int k0, int buf) {
        #pragma unroll
        for (int i = 0; i < 2; ++i) {
            const unsigned short* ga = &A [(size_t)(m0 + r0 + i * 64) * K + k0 + off];
            const unsigned short* gb = &Bt[(size_t)(n0 + r0 + i * 64) * K + k0 + off];
            async_b128(lds_addr(&As[buf][(r0 + i * 64) * LDS_PITCH + off]), ga);
            async_b128(lds_addr(&Bs[buf][(r0 + i * 64) * LDS_PITCH + off]), gb);
        }
    };

    union FragU { uint4 q[2]; v16bf v; };

    v8f acc[2][4] = {};

    issueTile(0, 0);

    // ISA 16-bit fragment layouts (wave32):
    //  A (16x32): lane<16 holds K{0..7}+{16..23}; lane>=16 holds K{8..15}+{24..31}
    //  B (32x16): lane<16 holds K{0..7}+{16..23}; lane>=16 holds K{8..15}+{24..31}
    const int aBase = (lane & 16) ? 8  : 0;
    const int bBase = (lane & 16) ? 8  : 0;
    const int aRow0 = wm * 32 + (lane & 15);
    const int bRow0 = wn * 64 + (lane & 15);

    const int nk = K >> 5;
    for (int kt = 0; kt < nk; ++kt) {
        wait_async0();        // my copies for tile kt are done
        __syncthreads();      // everyone's copies are done; prev reads retired

        if (kt + 1 < nk) issueTile((kt + 1) << 5, (kt + 1) & 1);  // overlap DMA

        const int buf = kt & 1;
        FragU af[2], bfg[4];
        #pragma unroll
        for (int mi = 0; mi < 2; ++mi) {
            int base = (aRow0 + mi * 16) * LDS_PITCH + aBase;
            af[mi].q[0] = *(const uint4*)&As[buf][base];
            af[mi].q[1] = *(const uint4*)&As[buf][base + 16];
        }
        #pragma unroll
        for (int ni = 0; ni < 4; ++ni) {
            int base = (bRow0 + ni * 16) * LDS_PITCH + bBase;
            bfg[ni].q[0] = *(const uint4*)&Bs[buf][base];
            bfg[ni].q[1] = *(const uint4*)&Bs[buf][base + 16];
        }
        #pragma unroll
        for (int mi = 0; mi < 2; ++mi)
            #pragma unroll
            for (int ni = 0; ni < 4; ++ni)
                acc[mi][ni] = __builtin_amdgcn_wmma_f32_16x16x32_bf16(
                    false, af[mi].v, false, bfg[ni].v,
                    (short)0, acc[mi][ni], false, false);
    }

    // Epilogue. C layout: VGPR r, lane<16 -> (M=r, N=lane); lane>=16 -> (M=r+8)
    const int nColBase = n0 + wn * 64 + (lane & 15);
    const int mRowBase = m0 + wm * 32 + ((lane & 16) ? 8 : 0);
    #pragma unroll
    for (int ni = 0; ni < 4; ++ni) {
        int n = nColBase + ni * 16;
        float bv = bias[n];
        #pragma unroll
        for (int mi = 0; mi < 2; ++mi) {
            #pragma unroll
            for (int r = 0; r < 8; ++r) {
                float v = acc[mi][ni][r] + bv;
                if (RELU) v = fmaxf(v, 0.f);
                size_t idx = (size_t)(mRowBase + mi * 16 + r) * N + n;
                if (OUT_BF16) ((unsigned short*)Out)[idx] = f2bf(v);
                else          ((float*)Out)[idx]          = v;
            }
        }
    }
}

// ----------------------------- mixing kernel -------------------------------
// Per sample b: hidden = ELU(emb[32x64] @ w1[64x32] + b1); out = hidden @ w2
// + b2; norm = min(1, 5/(max_v|out|+1e-5)); sigmoid(out*norm).
// All per-sample operands staged to LDS via async DMA.

__global__ __launch_bounds__(256)
void mix_kernel(const float* __restrict__ emb,
                const float* __restrict__ W1, const float* __restrict__ B1,
                const float* __restrict__ W2, const float* __restrict__ B2,
                float* __restrict__ out) {
    __shared__ __align__(16) float sm[9216];   // w1|b1|w2|b2|emb
    __shared__ float hid[32 * 32];
    __shared__ float os [32 * 64];
    __shared__ float nrm[64];

    float* w1s = sm;            // 2048  [a][h]
    float* b1s = sm + 2048;     // 1024  [v][h]
    float* w2s = sm + 3072;     // 2048  [h][f]
    float* b2s = sm + 5120;     // 2048  [v][f]
    float* es  = sm + 7168;     // 2048  [v][a]

    const int b = blockIdx.x;
    const int t = threadIdx.x;

    auto cp = [&](float* dst, const float* src, int nfloats) {
        int chunks = nfloats >> 2;                      // 16B chunks
        for (int i = t; i < chunks; i += 256)
            async_b128(lds_addr(&dst[i * 4]), src + i * 4);
    };
    cp(w1s, W1 + (size_t)b * 2048, 2048);
    cp(b1s, B1 + (size_t)b * 1024, 1024);
    cp(w2s, W2 + (size_t)b * 2048, 2048);
    cp(b2s, B2 + (size_t)b * 2048, 2048);
    cp(es,  emb + (size_t)b * 2048, 2048);
    wait_async0();
    __syncthreads();

    for (int i = t; i < 1024; i += 256) {          // hidden[v][h]
        int v = i >> 5, h = i & 31;
        float s = b1s[i];
        #pragma unroll 8
        for (int a = 0; a < 64; ++a) s += es[v * 64 + a] * w1s[a * 32 + h];
        hid[i] = (s > 0.f) ? s : expm1f(s);        // ELU
    }
    __syncthreads();

    for (int i = t; i < 2048; i += 256) {          // out[v][f]
        int v = i >> 6, f = i & 63;
        float s = b2s[i];
        #pragma unroll 8
        for (int h = 0; h < 32; ++h) s += hid[v * 32 + h] * w2s[h * 64 + f];
        os[i] = s;
    }
    __syncthreads();

    if (t < 64) {                                  // abs-max over v
        float m = 0.f;
        #pragma unroll 8
        for (int v = 0; v < 32; ++v) m = fmaxf(m, fabsf(os[v * 64 + t]));
        m += 1e-5f;
        nrm[t] = fminf(1.f, 5.f / m);
    }
    __syncthreads();

    float* op = out + (size_t)b * 2048;
    for (int i = t; i < 2048; i += 256) {
        float x = os[i] * nrm[i & 63];
        op[i] = 1.f / (1.f + expf(-x));
    }
}

// ------------------------------ launcher -----------------------------------

extern "C" void kernel_launch(void* const* d_in, const int* in_sizes, int n_in,
                              void* d_out, int out_size, void* d_ws, size_t ws_size,
                              hipStream_t stream) {
    (void)in_sizes; (void)n_in; (void)out_size; (void)ws_size;

    const float* agent_emb = (const float*)d_in[0];
    const float* states    = (const float*)d_in[1];
    const float* w1a_W = (const float*)d_in[2];  const float* w1a_b = (const float*)d_in[3];
    const float* w1b_W = (const float*)d_in[4];  const float* w1b_b = (const float*)d_in[5];
    const float* b1_W  = (const float*)d_in[6];  const float* b1_b  = (const float*)d_in[7];
    const float* w2a_W = (const float*)d_in[8];  const float* w2a_b = (const float*)d_in[9];
    const float* w2b_W = (const float*)d_in[10]; const float* w2b_b = (const float*)d_in[11];
    const float* b2a_W = (const float*)d_in[12]; const float* b2a_b = (const float*)d_in[13];
    const float* b2b_W = (const float*)d_in[14]; const float* b2b_b = (const float*)d_in[15];

    char* p = (char*)d_ws;
    auto carve = [&](size_t bytes) -> void* {
        void* q = (void*)p;
        p += (bytes + 255) & ~(size_t)255;
        return q;
    };

    unsigned short* Sb   = (unsigned short*)carve((size_t)B_SZ * SD_SZ * 2);
    unsigned short* w1aT = (unsigned short*)carve((size_t)H_SZ * SD_SZ * 2);       // [2048][512]
    unsigned short* w1bT = (unsigned short*)carve((size_t)H_SZ * H_SZ * 2);        // [2048][2048]
    unsigned short* b1T  = (unsigned short*)carve((size_t)1024 * SD_SZ * 2);       // [1024][512]
    unsigned short* w2aT = (unsigned short*)carve((size_t)4096 * SD_SZ * 2);       // [4096][512]
    unsigned short* w2bT = (unsigned short*)carve((size_t)2048 * 4096 * 2);        // [2048][4096]
    unsigned short* b2aT = (unsigned short*)carve((size_t)1024 * SD_SZ * 2);       // [1024][512]
    unsigned short* b2bT = (unsigned short*)carve((size_t)2048 * 1024 * 2);        // [2048][1024]
    unsigned short* H1b  = (unsigned short*)carve((size_t)B_SZ * 2048 * 2);
    unsigned short* H2b  = (unsigned short*)carve((size_t)B_SZ * 4096 * 2);
    unsigned short* HB2b = (unsigned short*)carve((size_t)B_SZ * 1024 * 2);
    float* W1f = (float*)carve((size_t)B_SZ * 2048 * 4);
    float* B1f = (float*)carve((size_t)B_SZ * 1024 * 4);
    float* W2f = (float*)carve((size_t)B_SZ * 2048 * 4);
    float* B2f = (float*)carve((size_t)B_SZ * 2048 * 4);

    dim3 blk(256);

    // ---- precision conversion / weight transposition (bf16) ----
    cvt_bf16_kernel<<<(B_SZ * SD_SZ + 255) / 256, blk, 0, stream>>>(states, Sb, B_SZ * SD_SZ);
    cvt_transpose_kernel<<<(SD_SZ * H_SZ   + 255) / 256, blk, 0, stream>>>(w1a_W, w1aT, SD_SZ, H_SZ);
    cvt_transpose_kernel<<<(H_SZ  * H_SZ   + 255) / 256, blk, 0, stream>>>(w1b_W, w1bT, H_SZ,  H_SZ);
    cvt_transpose_kernel<<<(SD_SZ * 1024   + 255) / 256, blk, 0, stream>>>(b1_W,  b1T,  SD_SZ, 1024);
    cvt_transpose_kernel<<<(SD_SZ * 4096   + 255) / 256, blk, 0, stream>>>(w2a_W, w2aT, SD_SZ, 4096);
    cvt_transpose_kernel<<<(4096  * 2048   + 255) / 256, blk, 0, stream>>>(w2b_W, w2bT, 4096,  2048);
    cvt_transpose_kernel<<<(SD_SZ * 1024   + 255) / 256, blk, 0, stream>>>(b2a_W, b2aT, SD_SZ, 1024);
    cvt_transpose_kernel<<<(1024  * 2048   + 255) / 256, blk, 0, stream>>>(b2b_W, b2bT, 1024,  2048);

    // ---- hypernet GEMMs (WMMA bf16 -> f32) ----
    const int MT = B_SZ / 128;  // 128 m-tiles
    gemm_bf16_wmma<true,  true ><<<dim3(2048/128, MT), blk, 0, stream>>>(Sb,   w1aT, w1a_b, H1b,  2048, 512);
    gemm_bf16_wmma<false, false><<<dim3(2048/128, MT), blk, 0, stream>>>(H1b,  w1bT, w1b_b, W1f,  2048, 2048);
    gemm_bf16_wmma<false, false><<<dim3(1024/128, MT), blk, 0, stream>>>(Sb,   b1T,  b1_b,  B1f,  1024, 512);
    gemm_bf16_wmma<true,  true ><<<dim3(4096/128, MT), blk, 0, stream>>>(Sb,   w2aT, w2a_b, H2b,  4096, 512);
    gemm_bf16_wmma<false, false><<<dim3(2048/128, MT), blk, 0, stream>>>(H2b,  w2bT, w2b_b, W2f,  2048, 4096);
    gemm_bf16_wmma<true,  true ><<<dim3(1024/128, MT), blk, 0, stream>>>(Sb,   b2aT, b2a_b, HB2b, 1024, 512);
    gemm_bf16_wmma<false, false><<<dim3(2048/128, MT), blk, 0, stream>>>(HB2b, b2bT, b2b_b, B2f,  2048, 1024);

    // ---- per-sample mixing + normalization + sigmoid ----
    mix_kernel<<<B_SZ, blk, 0, stream>>>(agent_emb, W1f, B1f, W2f, B2f, (float*)d_out);
}
